// Encoder_14654428414106
// MI455X (gfx1250) — matrix operand order repaired and checked by reading, not compiled, not run on hardware
//
#include <hip/hip_runtime.h>
#include <hip/hip_bf16.h>
#include <float.h>
#include <math.h>

// ---------------------------------------------------------------------------
// CDNA5 (gfx1250) implementation. wave32. WMMA f16 (16x16x32, f32 accum) for
// every GEMM; FPS / ball-query are latency-bound scalar/VALU kernels.
// ---------------------------------------------------------------------------

typedef __attribute__((ext_vector_type(16))) _Float16 v16h;
typedef __attribute__((ext_vector_type(8)))  _Float16 v8h;
typedef __attribute__((ext_vector_type(8)))  float    v8f;

#define BN_SC 0.99999500003749972f   // 1/sqrt(1+1e-5)

__device__ __forceinline__ int fenc(float f){            // monotone float->int
  int i = __float_as_int(f);
  return i ^ ((i >> 31) & 0x7fffffff);
}
__device__ __forceinline__ float fdec(int e){
  return __int_as_float(e ^ ((e >> 31) & 0x7fffffff));
}

__device__ __forceinline__ v8f wmma32(v16h a, v16h b, v8f c){
  // v_wmma_f32_16x16x32_f16  (doc-verified 8-arg signature)
  return __builtin_amdgcn_wmma_f32_16x16x32_f16(false, a, false, b, (short)0, c,
                                                false, false);
}

// A fragment (16x32 f16, M x K). row = base + (lane&15)*stride, caller-computed.
// lane<16: K = {k0..k0+7, k0+16..k0+23} with k0 = 32*kstep;
// lane>=16: same pattern shifted by 8.
__device__ __forceinline__ v16h load_a_frag(const _Float16* row, int kstep){
  const int lane = threadIdx.x & 31;
  const int k0 = kstep * 32 + ((lane & 16) ? 8 : 0);
  v8h lo = *(const v8h*)(row + k0);
  v8h hi = *(const v8h*)(row + k0 + 16);
  v16h a;
#pragma unroll
  for (int i = 0; i < 8; ++i){ a[i] = lo[i]; a[i + 8] = hi[i]; }
  return a;
}

// B fragment (32x16 f16, K x N) from weights stored row-major [Npad][Kpad]
// (i.e. B[k][n] = W[n][k]); same K pattern as load_a_frag (two 8-half runs).
__device__ __forceinline__ v16h load_b_frag(const _Float16* W, int Kpad,
                                            int ntile, int kstep){
  const int lane = threadIdx.x & 31;
  const _Float16* p = W + (size_t)(ntile * 16 + (lane & 15)) * Kpad
                        + kstep * 32 + ((lane & 16) ? 8 : 0);
  v8h lo = *(const v8h*)p;
  v8h hi = *(const v8h*)(p + 16);
  v16h b;
#pragma unroll
  for (int i = 0; i < 8; ++i){ b[i] = lo[i]; b[i + 8] = hi[i]; }
  return b;
}

// ------------------------------- weight prep -------------------------------
__global__ void k_cvt_pad(const float* __restrict__ src, _Float16* __restrict__ dst,
                          int r, int c, int rp, int cp){
  int i = blockIdx.x * blockDim.x + threadIdx.x;
  int total = rp * cp;
  if (i < total){
    int rr = i / cp, cc = i - rr * cp;
    dst[i] = (_Float16)((rr < r && cc < c) ? src[(size_t)rr * c + cc] : 0.f);
  }
}

__global__ void k_zinit(int* __restrict__ z, int n){
  int i = blockIdx.x * blockDim.x + threadIdx.x;
  if (i < n) z[i] = fenc(-FLT_MAX);
}

// ----------------------------------- FPS -----------------------------------
// One WG per batch element; dists live in LDS; 960 total dependent rounds are
// the serial critical path of the whole net, so keep them WGP-resident.
__global__ void __launch_bounds__(256)
k_fps(const float* __restrict__ xyz, int N, int np,
      int* __restrict__ sidx, float* __restrict__ new_xyz){
  extern __shared__ float sm[];
  float* dists = sm;                 // N
  float* rv    = sm + N;             // 256
  int*   ri    = (int*)(rv + 256);   // 256
  __shared__ float cent[3];
  __shared__ int   s_far;
  const int b = blockIdx.x;
  const float* xb = xyz + (size_t)b * N * 3;
  for (int j = threadIdx.x; j < N; j += 256) dists[j] = 1e10f;
  int far = 0;
  __syncthreads();
  for (int it = 0; it < np; ++it){
    if (threadIdx.x == 0){
      float cx = xb[(size_t)far*3+0], cy = xb[(size_t)far*3+1], cz = xb[(size_t)far*3+2];
      cent[0] = cx; cent[1] = cy; cent[2] = cz;
      sidx[(size_t)b*np + it] = far;
      float* o = new_xyz + ((size_t)b*np + it) * 3;
      o[0] = cx; o[1] = cy; o[2] = cz;
    }
    __syncthreads();
    const float cx = cent[0], cy = cent[1], cz = cent[2];
    float best = -1.f; int bidx = 0;
    for (int j = threadIdx.x; j < N; j += 256){
      float dx = xb[(size_t)j*3+0]-cx, dy = xb[(size_t)j*3+1]-cy, dz = xb[(size_t)j*3+2]-cz;
      float d = fminf(dists[j], dx*dx + dy*dy + dz*dz);
      dists[j] = d;
      if (d > best){ best = d; bidx = j; }   // strided ascending j => first-max kept
    }
    rv[threadIdx.x] = best; ri[threadIdx.x] = bidx;
    __syncthreads();
    for (int s = 128; s > 0; s >>= 1){
      if (threadIdx.x < s){
        float ov = rv[threadIdx.x + s]; int oi = ri[threadIdx.x + s];
        if (ov > rv[threadIdx.x] ||
            (ov == rv[threadIdx.x] && oi < ri[threadIdx.x])){
          rv[threadIdx.x] = ov; ri[threadIdx.x] = oi;
        }
      }
      __syncthreads();
    }
    if (threadIdx.x == 0) s_far = ri[0];
    __syncthreads();
    far = s_far;
  }
}

// -------------------------------- ball query -------------------------------
// One wave32 per query: ballot + prefix-popcount compaction emits the first
// 64 in-radius indices in index order (== reference's sorted-key select).
__global__ void __launch_bounds__(32)
k_ballq(const float* __restrict__ xyz, const float* __restrict__ new_xyz,
        int N, int np, float r2, int* __restrict__ nidx){
  const int q = blockIdx.x;
  const int b = q / np;
  const int lane = threadIdx.x;
  const float* xb = xyz + (size_t)b * N * 3;
  const float qx = new_xyz[(size_t)q*3+0];
  const float qy = new_xyz[(size_t)q*3+1];
  const float qz = new_xyz[(size_t)q*3+2];
  int* out = nidx + (size_t)q * 64;
  int cnt = 0, first = -1;
  for (int base = 0; base < N && cnt < 64; base += 32){
    int j = base + lane;
    bool in = false;
    if (j < N){
      float dx = xb[(size_t)j*3+0]-qx, dy = xb[(size_t)j*3+1]-qy, dz = xb[(size_t)j*3+2]-qz;
      in = (dx*dx + dy*dy + dz*dz) < r2;
    }
    unsigned m = (unsigned)__ballot(in);     // wave32: low 32 bits
    if (first < 0 && m != 0u) first = base + __ffs((int)m) - 1;
    if (in){
      int pos = cnt + __popc(m & ((1u << lane) - 1u));
      if (pos < 64) out[pos] = j;
    }
    cnt += __popc(m);
  }
  if (cnt > 64) cnt = 64;
  int fill = (first >= 0) ? first : (N - 1);  // none found -> N-1 (ref semantics)
  for (int k = cnt + lane; k < 64; k += 32) out[k] = fill;
}

// ------------------------- fused set-abstraction stage ----------------------
// One WG (128 thr = 4 waves) per query: 64 samples = 4 row strips of 16.
// hfeat(10,pad32) --WMMA--> h1(64) --WMMA--> chid, fused bn/relu/gate/max,
// then chid->cout projection (tiny, VALU).
__global__ void __launch_bounds__(128)
k_stage(const float* __restrict__ xyz, const float* __restrict__ feats_in,
        const float* __restrict__ new_xyz, const int* __restrict__ nidx,
        const _Float16* __restrict__ m1h, const _Float16* __restrict__ m2h,
        const float* __restrict__ m1_b,  const float* __restrict__ m2_b,
        const float* __restrict__ bnm_g, const float* __restrict__ bnm_b,
        const float* __restrict__ bnr_g, const float* __restrict__ bnr_b,
        const float* __restrict__ cr_w,  const float* __restrict__ cr_b,
        const float* __restrict__ bnc_g, const float* __restrict__ bnc_b,
        const float* __restrict__ xr_w,  const float* __restrict__ xr_b,
        const float* __restrict__ bnx_g, const float* __restrict__ bnx_b,
        float* __restrict__ feats_out,
        int N, int np, int chid, int chid_pad, int cout, int cprev, int first){
  __shared__ _Float16 s_hf[64 * 32];
  __shared__ _Float16 s_h1[64 * 64];
  __shared__ float    s_rel[64 * 3];
  __shared__ int      s_gidx[64];
  __shared__ float    s_xf[64 * 16];
  __shared__ int      s_ymax[272];

  const int q = blockIdx.x;
  const int b = q / np;
  const int tid = threadIdx.x, wave = tid >> 5, lane = tid & 31;
  const int ns = 64;

  const float nx = new_xyz[(size_t)q*3+0];
  const float ny = new_xyz[(size_t)q*3+1];
  const float nz = new_xyz[(size_t)q*3+2];

  for (int m = tid; m < ns; m += 128){
    int j = nidx[(size_t)q*ns + m];
    int g = b * N + j;
    s_gidx[m] = g;
    s_rel[m*3+0] = xyz[(size_t)g*3+0] - nx;
    s_rel[m*3+1] = xyz[(size_t)g*3+1] - ny;
    s_rel[m*3+2] = xyz[(size_t)g*3+2] - nz;
  }
  for (int t = tid; t < chid_pad; t += 128) s_ymax[t] = 0;  // relu output >= 0
  __syncthreads();

  const float g0x = s_rel[0] + nx, g0y = s_rel[1] + ny, g0z = s_rel[2] + nz;
  for (int m = tid; m < ns; m += 128){
    float rx = s_rel[m*3+0], ry = s_rel[m*3+1], rz = s_rel[m*3+2];
    float hd = sqrtf(rx*rx + ry*ry + rz*rz + 1e-12f);
    _Float16* hr = s_hf + m * 32;
    hr[0] = (_Float16)hd;
    hr[1] = (_Float16)g0x;      hr[2] = (_Float16)g0y;      hr[3] = (_Float16)g0z;
    hr[4] = (_Float16)(rx+nx);  hr[5] = (_Float16)(ry+ny);  hr[6] = (_Float16)(rz+nz);
    hr[7] = (_Float16)rx;       hr[8] = (_Float16)ry;       hr[9] = (_Float16)rz;
#pragma unroll
    for (int k = 10; k < 32; ++k) hr[k] = (_Float16)0.f;
  }
  if (first){
    for (int t = tid; t < ns * 16; t += 128){   // xr: 3->16, bn, relu (tiny)
      int m = t >> 4, c = t & 15;
      float a = xr_b[c] + s_rel[m*3+0]*xr_w[c*3+0]
                        + s_rel[m*3+1]*xr_w[c*3+1]
                        + s_rel[m*3+2]*xr_w[c*3+2];
      a = a * (bnx_g[c]*BN_SC) + bnx_b[c];
      s_xf[t] = fmaxf(a, 0.f);
    }
  }
  __syncthreads();

  { // GEMM1: [64x32] x [32x64]  (K padded 10->32)
    const _Float16* arow = s_hf + (size_t)(wave*16 + (lane & 15)) * 32;
    v16h a = load_a_frag(arow, 0);
#pragma unroll
    for (int t = 0; t < 4; ++t){
      v8f c = {};
      c = wmma32(a, load_b_frag(m1h, 32, t, 0), c);
      const int n = t*16 + (lane & 15);
      const float bias = m1_b[n];
#pragma unroll
      for (int r = 0; r < 8; ++r){
        int m = wave*16 + r + ((lane & 16) ? 8 : 0);
        s_h1[m*64 + n] = (_Float16)fmaxf(c[r] + bias, 0.f);
      }
    }
  }
  __syncthreads();

  { // GEMM2: [64x64] x [64xchid_pad], fused bn/relu * gate * bn/relu, max
    const _Float16* arow = s_h1 + (size_t)(wave*16 + (lane & 15)) * 64;
    v16h a0 = load_a_frag(arow, 0);
    v16h a1 = load_a_frag(arow, 1);
    const int ntiles = chid_pad >> 4;
    for (int t = 0; t < ntiles; ++t){
      v8f c = {};
      c = wmma32(a0, load_b_frag(m2h, 64, t, 0), c);
      c = wmma32(a1, load_b_frag(m2h, 64, t, 1), c);
      const int n = t*16 + (lane & 15);
      if (n < chid){
        const float bb = m2_b[n];
        const float gm = bnm_g[n]*BN_SC, bm = bnm_b[n];
        const float gr = bnr_g[n]*BN_SC, br = bnr_b[n];
        float lmax = 0.f;
#pragma unroll
        for (int r = 0; r < 8; ++r){
          int m = wave*16 + r + ((lane & 16) ? 8 : 0);
          float h = fmaxf((c[r] + bb)*gm + bm, 0.f);
          float xv;
          if (first)      xv = s_xf[m*16 + n];
          else if (n < 3) xv = s_rel[m*3 + n];
          else            xv = feats_in[(size_t)s_gidx[m]*cprev + (n - 3)];
          float y = fmaxf((h * xv)*gr + br, 0.f);
          lmax = fmaxf(lmax, y);
        }
        atomicMax(&s_ymax[n], __float_as_int(lmax));  // bits monotone for >=0
      }
    }
  }
  __syncthreads();

  for (int c0 = tid; c0 < cout; c0 += 128){  // cr: chid -> cout, bn, relu
    float acc = cr_b[c0];
    const float* wr = cr_w + (size_t)c0 * chid;
    for (int k = 0; k < chid; ++k) acc += __int_as_float(s_ymax[k]) * wr[k];
    acc = acc * (bnc_g[c0]*BN_SC) + bnc_b[c0];
    feats_out[(size_t)q*cout + c0] = fmaxf(acc, 0.f);
  }
}

// --------------------------------- conv_sym --------------------------------
// 32 rows per WG (128 thr); wave -> (row-strip = wave&1, N-tiles = wave>>1::2).
// cin->256->256->64 all via WMMA; global max via encoded-int atomicMax.
__global__ void __launch_bounds__(128)
k_convsym(const float* __restrict__ feats,
          const _Float16* __restrict__ w1h, const _Float16* __restrict__ w2h,
          const _Float16* __restrict__ w3h,
          const float* __restrict__ b1, const float* __restrict__ g1, const float* __restrict__ be1,
          const float* __restrict__ b2, const float* __restrict__ g2, const float* __restrict__ be2,
          const float* __restrict__ b3,
          int np, int cin, int* __restrict__ zbuf){
  __shared__ _Float16 bufA[32 * 256];
  __shared__ _Float16 bufB[32 * 256];
  const int tid = threadIdx.x, wave = tid >> 5, lane = tid & 31;
  const int b = blockIdx.y;
  const int row0 = blockIdx.x * 32;
  const float* fb = feats + ((size_t)b*np + row0) * cin;

  __builtin_prefetch((const void*)w2h, 0, 1);   // global_prefetch of next layer

  for (int t = tid; t < 32 * 256; t += 128){
    int m = t >> 8, c = t & 255;
    bufA[t] = (_Float16)((c < cin) ? fb[(size_t)m*cin + c] : 0.f);
  }
  __syncthreads();

  const int mtile  = wave & 1;
  const int tstart = wave >> 1;
  const _Float16* arowA = bufA + (size_t)(mtile*16 + (lane & 15)) * 256;
  const _Float16* arowB = bufB + (size_t)(mtile*16 + (lane & 15)) * 256;

  for (int t = tstart; t < 16; t += 2){          // L1: cin(pad256) -> 256
    v8f c = {};
#pragma unroll
    for (int k = 0; k < 8; ++k)
      c = wmma32(load_a_frag(arowA, k), load_b_frag(w1h, 256, t, k), c);
    const int n = t*16 + (lane & 15);
    const float sc = g1[n]*BN_SC, bi = b1[n], sh = be1[n];
#pragma unroll
    for (int r = 0; r < 8; ++r){
      int m = mtile*16 + r + ((lane & 16) ? 8 : 0);
      bufB[m*256 + n] = (_Float16)fmaxf((c[r] + bi)*sc + sh, 0.f);
    }
  }
  __syncthreads();
  for (int t = tstart; t < 16; t += 2){          // L2: 256 -> 256
    v8f c = {};
#pragma unroll
    for (int k = 0; k < 8; ++k)
      c = wmma32(load_a_frag(arowB, k), load_b_frag(w2h, 256, t, k), c);
    const int n = t*16 + (lane & 15);
    const float sc = g2[n]*BN_SC, bi = b2[n], sh = be2[n];
#pragma unroll
    for (int r = 0; r < 8; ++r){
      int m = mtile*16 + r + ((lane & 16) ? 8 : 0);
      bufA[m*256 + n] = (_Float16)fmaxf((c[r] + bi)*sc + sh, 0.f);
    }
  }
  __syncthreads();
  for (int t = tstart; t < 4; t += 2){           // L3: 256 -> 64, max over rows
    v8f c = {};
#pragma unroll
    for (int k = 0; k < 8; ++k)
      c = wmma32(load_a_frag(arowA, k), load_b_frag(w3h, 256, t, k), c);
    const int n = t*16 + (lane & 15);
    float lmax = -FLT_MAX;
#pragma unroll
    for (int r = 0; r < 8; ++r) lmax = fmaxf(lmax, c[r] + b3[n]);
    atomicMax(&zbuf[b*64 + n], fenc(lmax));      // signed-safe encoded max
  }
}

__global__ void k_final(const int* __restrict__ zbuf, float* __restrict__ out){
  int t = blockIdx.x * blockDim.x + threadIdx.x;
  if (t < 16 * 256){
    int b = t >> 8, c = t & 255;
    int s = c >> 6, j = c & 63;
    out[t] = fdec(zbuf[(s*16 + b)*64 + j]);
  }
}

// --------------------------------- launch ----------------------------------
extern "C" void kernel_launch(void* const* d_in, const int* in_sizes, int n_in,
                              void* d_out, int out_size, void* d_ws, size_t ws_size,
                              hipStream_t stream){
  (void)in_sizes; (void)n_in; (void)out_size; (void)ws_size;
  const int Bn = 16;
  static const int   NP[4]    = {512, 256, 128, 64};
  static const float RAD[4]   = {0.1f, 0.2f, 0.3f, 0.4f};
  static const int   CHID[4]  = {16, 131, 259, 259};
  static const int   CHIDP[4] = {16, 144, 272, 272};
  static const int   COUT[4]  = {128, 256, 256, 256};
  static const int   CINCS[4] = {128, 256, 256, 256};

  // d_in[0] = pointcloud; then params flattened as jax tree leaves
  // (sorted dict keys): 'cs'[0..3]{b1,b2,b3,be1,be2,g1,g2,w1,w2,w3},
  // then 'sa'[0..3] with sorted keys (stage0 has the 4 extra bnx/xr leaves).
  const float* pc = (const float*)d_in[0];

  struct CSP { const float *b1,*b2,*b3,*be1,*be2,*g1,*g2,*w1,*w2,*w3; } cs[4];
  for (int s = 0; s < 4; ++s){
    int base = 1 + s*10;
    cs[s].b1  = (const float*)d_in[base+0];
    cs[s].b2  = (const float*)d_in[base+1];
    cs[s].b3  = (const float*)d_in[base+2];
    cs[s].be1 = (const float*)d_in[base+3];
    cs[s].be2 = (const float*)d_in[base+4];
    cs[s].g1  = (const float*)d_in[base+5];
    cs[s].g2  = (const float*)d_in[base+6];
    cs[s].w1  = (const float*)d_in[base+7];
    cs[s].w2  = (const float*)d_in[base+8];
    cs[s].w3  = (const float*)d_in[base+9];
  }
  struct SAP { const float *bnc_b,*bnc_g,*bnm_b,*bnm_g,*bnr_b,*bnr_g,
               *bnx_b,*bnx_g,*cr_b,*cr_w,*m1_b,*m1_w,*m2_b,*m2_w,*xr_b,*xr_w; } sa[4] = {};
  {
    int p = 41;
    sa[0].bnc_b=(const float*)d_in[p++]; sa[0].bnc_g=(const float*)d_in[p++];
    sa[0].bnm_b=(const float*)d_in[p++]; sa[0].bnm_g=(const float*)d_in[p++];
    sa[0].bnr_b=(const float*)d_in[p++]; sa[0].bnr_g=(const float*)d_in[p++];
    sa[0].bnx_b=(const float*)d_in[p++]; sa[0].bnx_g=(const float*)d_in[p++];
    sa[0].cr_b =(const float*)d_in[p++]; sa[0].cr_w =(const float*)d_in[p++];
    sa[0].m1_b =(const float*)d_in[p++]; sa[0].m1_w =(const float*)d_in[p++];
    sa[0].m2_b =(const float*)d_in[p++]; sa[0].m2_w =(const float*)d_in[p++];
    sa[0].xr_b =(const float*)d_in[p++]; sa[0].xr_w =(const float*)d_in[p++];
    for (int s = 1; s < 4; ++s){
      sa[s].bnc_b=(const float*)d_in[p++]; sa[s].bnc_g=(const float*)d_in[p++];
      sa[s].bnm_b=(const float*)d_in[p++]; sa[s].bnm_g=(const float*)d_in[p++];
      sa[s].bnr_b=(const float*)d_in[p++]; sa[s].bnr_g=(const float*)d_in[p++];
      sa[s].cr_b =(const float*)d_in[p++]; sa[s].cr_w =(const float*)d_in[p++];
      sa[s].m1_b =(const float*)d_in[p++]; sa[s].m1_w =(const float*)d_in[p++];
      sa[s].m2_b =(const float*)d_in[p++]; sa[s].m2_w =(const float*)d_in[p++];
    }
  }

  // ---- carve workspace ----
  char* wp = (char*)d_ws;
  auto carve = [&](size_t bytes) -> void* {
    void* r = (void*)wp;
    wp += (bytes + 255) & ~(size_t)255;
    return r;
  };
  _Float16 *m1h[4], *m2h[4], *w1h[4], *w2h[4], *w3h[4];
  int *sidx[4], *nidxp[4];
  float *nxyz[4], *featp[4];
  for (int s = 0; s < 4; ++s){
    m1h[s]  = (_Float16*)carve((size_t)64*32*2);
    m2h[s]  = (_Float16*)carve((size_t)CHIDP[s]*64*2);
    w1h[s]  = (_Float16*)carve((size_t)256*256*2);
    w2h[s]  = (_Float16*)carve((size_t)256*256*2);
    w3h[s]  = (_Float16*)carve((size_t)64*256*2);
    sidx[s] = (int*)  carve((size_t)Bn*NP[s]*4);
    nidxp[s]= (int*)  carve((size_t)Bn*NP[s]*64*4);
    nxyz[s] = (float*)carve((size_t)Bn*NP[s]*3*4);
    featp[s]= (float*)carve((size_t)Bn*NP[s]*COUT[s]*4);
  }
  int* zbuf = (int*)carve((size_t)4*Bn*64*4);

  // ---- convert weights to padded f16 (once per launch) ----
  auto cvt = [&](const float* src, _Float16* dst, int r, int c, int rp, int cp){
    int total = rp * cp;
    k_cvt_pad<<<(total + 255)/256, 256, 0, stream>>>(src, dst, r, c, rp, cp);
  };
  for (int s = 0; s < 4; ++s){
    cvt(sa[s].m1_w, m1h[s], 64, 10, 64, 32);
    cvt(sa[s].m2_w, m2h[s], CHID[s], 64, CHIDP[s], 64);
    cvt(cs[s].w1,   w1h[s], 256, CINCS[s], 256, 256);
    cvt(cs[s].w2,   w2h[s], 256, 256, 256, 256);
    cvt(cs[s].w3,   w3h[s], 64, 256, 64, 256);
  }
  k_zinit<<<(4*Bn*64 + 255)/256, 256, 0, stream>>>(zbuf, 4*Bn*64);

  // ---- stage chain ----
  const float* xyz = pc;
  const float* fin = nullptr;
  int N = 4096, cprev = 0;
  for (int s = 0; s < 4; ++s){
    const int np = NP[s];
    size_t lds = (size_t)N*4 + 256*4 + 256*4;
    k_fps<<<Bn, 256, lds, stream>>>(xyz, N, np, sidx[s], nxyz[s]);
    k_ballq<<<Bn*np, 32, 0, stream>>>(xyz, nxyz[s], N, np, RAD[s]*RAD[s], nidxp[s]);
    k_stage<<<Bn*np, 128, 0, stream>>>(xyz, fin, nxyz[s], nidxp[s], m1h[s], m2h[s],
        sa[s].m1_b, sa[s].m2_b, sa[s].bnm_g, sa[s].bnm_b, sa[s].bnr_g, sa[s].bnr_b,
        sa[s].cr_w, sa[s].cr_b, sa[s].bnc_g, sa[s].bnc_b,
        sa[s].xr_w, sa[s].xr_b, sa[s].bnx_g, sa[s].bnx_b,
        featp[s], N, np, CHID[s], CHIDP[s], COUT[s], cprev, s == 0 ? 1 : 0);
    k_convsym<<<dim3(np/32, Bn), 128, 0, stream>>>(featp[s], w1h[s], w2h[s], w3h[s],
        cs[s].b1, cs[s].g1, cs[s].be1, cs[s].b2, cs[s].g2, cs[s].be2, cs[s].b3,
        np, CINCS[s], zbuf + s*Bn*64);
    xyz = nxyz[s]; N = np; fin = featp[s]; cprev = COUT[s];
  }
  k_final<<<(Bn*256 + 255)/256, 256, 0, stream>>>(zbuf, (float*)d_out);
}
